// TraitsPredictor_66778151518675
// MI455X (gfx1250) — matrix operand order repaired and checked by reading, not compiled, not run on hardware
//
#include <hip/hip_runtime.h>

#define HDIM 128
#define NSP 50000
#define NSPEC 20000
#define ESS 800000
#define EBP 400000
#define EPP 320000

typedef __attribute__((ext_vector_type(16))) _Float16 v16h;
typedef __attribute__((ext_vector_type(8)))  _Float16 v8h;
typedef __attribute__((ext_vector_type(8)))  float    v8f;

// ---------- helpers ----------
__device__ __forceinline__ unsigned ordf(float f) {
  unsigned u = __float_as_uint(f);
  return (u & 0x80000000u) ? ~u : (u | 0x80000000u);
}
__device__ __forceinline__ float unordf(unsigned u) {
  unsigned b = (u & 0x80000000u) ? (u & 0x7FFFFFFFu) : ~u;
  return __uint_as_float(b);
}

// ---------- generic fill ----------
__global__ void k_fill_u32(unsigned* p, unsigned v, long n) {
  long i = (long)blockIdx.x * blockDim.x + threadIdx.x;
  if (i < n) p[i] = v;
}

// ---------- concat (up to 3 f32 sources) -> zero-padded f16 matrix ----------
__global__ void k_concat_f16(const float* a, int wa, const float* b, int wb,
                             const float* c, int wc, _Float16* out, int Kp, long total) {
  long i = (long)blockIdx.x * blockDim.x + threadIdx.x;
  if (i >= total) return;
  int k = (int)(i % Kp);
  long row = i / Kp;
  float v = 0.0f;
  if (k < wa)                 v = a[row * wa + k];
  else if (k < wa + wb)       v = b[row * wb + (k - wa)];
  else if (k < wa + wb + wc)  v = c[row * wc + (k - wa - wb)];
  out[i] = (_Float16)v;
}

// ---------- WMMA GEMM: C[M,N] = A[M,K] * B[N,K]^T  (f16 in, f32 out) ----------
// one wave per 16 x (TILES*16) output strip; K mult of 32, M mult of 16, N = gridDim.y*TILES*16
template <int TILES>
__global__ __launch_bounds__(32) void k_gemm_wmma(const _Float16* __restrict__ A,
                                                  const _Float16* __restrict__ B,
                                                  float* __restrict__ C,
                                                  int M, int N, int K) {
  int lane = threadIdx.x;
  int half = lane >> 4;       // 0: lanes 0-15, 1: lanes 16-31
  int l15  = lane & 15;
  int m0   = blockIdx.x * 16;
  int n0   = blockIdx.y * (TILES * 16);

  v8f acc[TILES];
#pragma unroll
  for (int t = 0; t < TILES; t++) acc[t] = (v8f){0.f,0.f,0.f,0.f,0.f,0.f,0.f,0.f};

  const _Float16* arow = A + (size_t)(m0 + l15) * K;
  const _Float16* brow[TILES];
#pragma unroll
  for (int t = 0; t < TILES; t++)
    brow[t] = B + (size_t)(n0 + t * 16 + l15) * K + half * 16;

  for (int k0 = 0; k0 < K; k0 += 32) {
    // A fragment: M = lane%16, K = (lane<16?0:16)+0..15 contiguous
    v16h a = *(const v16h*)(arow + k0 + half * 16);
#pragma unroll
    for (int t = 0; t < TILES; t++) {
      // B fragment: N = lane%16, K = (lane<16?0:16)+0..15 contiguous
      v16h b = *(const v16h*)(brow[t] + k0);
      acc[t] = __builtin_amdgcn_wmma_f32_16x16x32_f16(
          false, a, false, b, (short)0, acc[t], false, false);
    }
  }
#pragma unroll
  for (int t = 0; t < TILES; t++) {
    int n = n0 + t * 16 + l15;
#pragma unroll
    for (int r = 0; r < 8; r++)
      C[(size_t)(m0 + r + half * 8) * N + n] = acc[t][r];
  }
}

// ---------- per-row attention dots: a_src = h . att_src, a_dst = h . att_dst ----------
__global__ void k_row_dots(const float* __restrict__ h, const float* __restrict__ att_s,
                           const float* __restrict__ att_d, float* oas, float* oad, int M) {
  int wid  = (int)((blockIdx.x * (long)blockDim.x + threadIdx.x) >> 5);
  int lane = threadIdx.x & 31;
  if (wid >= M) return;
  const float* hr = h + (size_t)wid * HDIM;
  float s = 0.f, d = 0.f;
#pragma unroll
  for (int j = 0; j < 4; j++) {
    int k = lane + 32 * j;
    float hv = hr[k];
    s += hv * att_s[k];
    if (att_d) d += hv * att_d[k];
  }
  for (int o = 16; o > 0; o >>= 1) {
    s += __shfl_xor(s, o, 32);
    if (att_d) d += __shfl_xor(d, o, 32);
  }
  if (lane == 0) { oas[wid] = s; if (att_d) oad[wid] = d; }
}

// ---------- sum of edge attrs (for self-loop mean fill) ----------
__global__ void k_sum_attr(const float* __restrict__ a, int E, float* out) {
  int tid = blockIdx.x * blockDim.x + threadIdx.x;
  float s = 0.f;
  for (int i = tid; i < E; i += gridDim.x * blockDim.x) s += a[i];
  for (int o = 16; o > 0; o >>= 1) s += __shfl_xor(s, o, 32);
  if ((threadIdx.x & 31) == 0) atomicAdd(out, s);
}

// ---------- scal[1] = dot(lin_e, att_e); scal[2] = mean attr ----------
__global__ void k_layer_scalars(const float* lin_e, const float* att_e, float* scal, float invE) {
  __shared__ float red[4];
  int t = threadIdx.x;  // 128
  float v = lin_e[t] * att_e[t];
  for (int o = 16; o > 0; o >>= 1) v += __shfl_xor(v, o, 32);
  if ((t & 31) == 0) red[t >> 5] = v;
  __syncthreads();
  if (t == 0) {
    scal[1] = red[0] + red[1] + red[2] + red[3];
    scal[2] = scal[0] * invE;
  }
}

// ---------- pass 1: alpha = leaky(a_src[s] + c*attr + a_dst[d]); atomicMax per dst ----------
__global__ void k_edge_alpha_max(const int* __restrict__ src, const int* __restrict__ dst,
                                 const float* __restrict__ attr, int E, int nloops,
                                 const float* __restrict__ asrc, const float* __restrict__ adst,
                                 const float* __restrict__ scal,
                                 float* __restrict__ alpha, unsigned* __restrict__ mcode) {
  int i = blockIdx.x * blockDim.x + threadIdx.x;
  int total = E + nloops;
  if (i >= total) return;
  int sI, dI; float at;
  if (i < E) { sI = src[i]; dI = dst[i]; at = attr[i]; }
  else       { sI = dI = i - E; at = scal[2]; }   // self loop, mean attr
  float a = asrc[sI] + scal[1] * at;
  if (adst) a += adst[dI];
  a = (a > 0.f) ? a : 0.2f * a;                    // leaky_relu(0.2)
  alpha[i] = a;
  atomicMax(&mcode[dI], ordf(a));
}

// ---------- decode per-node max; non-finite (no edges) -> 0 ----------
__global__ void k_node_decode(const unsigned* mcode, float* mval, int n) {
  int i = blockIdx.x * blockDim.x + threadIdx.x;
  if (i >= n) return;
  float m = unordf(mcode[i]);
  unsigned e = (__float_as_uint(m) >> 23) & 0xFFu;
  mval[i] = (e == 0xFFu) ? 0.f : m;
}

// ---------- pass 2: e = exp(alpha - m[dst]); atomicAdd per dst (alpha overwritten) ----------
__global__ void k_edge_exp_sum(const int* __restrict__ dst, int E, int nloops,
                               const float* __restrict__ mval,
                               float* __restrict__ alpha, float* __restrict__ ssum) {
  int i = blockIdx.x * blockDim.x + threadIdx.x;
  int total = E + nloops;
  if (i >= total) return;
  int dI = (i < E) ? dst[i] : (i - E);
  float e = __expf(alpha[i] - mval[dI]);
  alpha[i] = e;
  atomicAdd(&ssum[dI], e);
}

// ---------- pass 3: agg[d] += (e/s[d]) * h[s], one wave per edge ----------
__global__ void k_edge_agg(const int* __restrict__ src, const int* __restrict__ dst,
                           int E, int nloops, const float* __restrict__ ev,
                           const float* __restrict__ ssum, const float* __restrict__ h,
                           float* __restrict__ agg) {
  int wid  = (int)((blockIdx.x * (long)blockDim.x + threadIdx.x) >> 5);
  int lane = threadIdx.x & 31;
  int total = E + nloops;
  if (wid >= total) return;
  int sI = (wid < E) ? src[wid] : (wid - E);
  int dI = (wid < E) ? dst[wid] : (wid - E);
  float w = ev[wid] / ssum[dI];
  const float* hs = h + (size_t)sI * HDIM;
  float* ag = agg + (size_t)dI * HDIM;
#pragma unroll
  for (int j = 0; j < 4; j++) {
    int k = lane + 32 * j;
    atomicAdd(&ag[k], w * hs[k]);
  }
}

// ---------- out = [relu](agg + bias) ----------
__global__ void k_finalize(const float* __restrict__ agg, const float* __restrict__ bias,
                           float* __restrict__ out, long total, int Hh, int relu) {
  long i = (long)blockIdx.x * blockDim.x + threadIdx.x;
  if (i >= total) return;
  float v = agg[i] + bias[i % Hh];
  if (relu) v = fmaxf(v, 0.f);
  out[i] = v;
}

// ================= host orchestration =================
static inline int ceil_div(long a, int b) { return (int)((a + b - 1) / b); }

struct WS {
  float *h_cur, *h_lin, *agg, *asrc, *adst, *alpha, *mval, *ssum, *scal;
  unsigned *mcode;
  _Float16 *xf16, *wf16;
};

static void gat_layer(hipStream_t s, const WS& W,
                      const float* xa, int wa, const float* xb, int wb,
                      const float* xc, int wc, int Msrc, int Kp,
                      const float* lin, const float* att_src, const float* att_dst,
                      const float* lin_e, const float* att_e, const float* bias,
                      const int* esrc, const int* edst, const float* eattr, int E,
                      int ndst, bool self_loops, bool relu_out, float* out) {
  long xtot = (long)Msrc * Kp;
  k_concat_f16<<<ceil_div(xtot, 256), 256, 0, s>>>(xa, wa, xb, wb, xc, wc, W.xf16, Kp, xtot);
  long wtot = (long)HDIM * Kp;
  k_concat_f16<<<ceil_div(wtot, 256), 256, 0, s>>>(lin, wa + wb + wc, nullptr, 0, nullptr, 0,
                                                   W.wf16, Kp, wtot);
  dim3 gg(Msrc / 16, HDIM / 64);  // N=128 -> grid.y = 2, 4 tiles per wave
  k_gemm_wmma<4><<<gg, 32, 0, s>>>(W.xf16, W.wf16, W.h_lin, Msrc, HDIM, Kp);
  k_row_dots<<<ceil_div(Msrc, 8), 256, 0, s>>>(W.h_lin, att_src, att_dst, W.asrc,
                                               att_dst ? W.adst : nullptr, Msrc);
  k_fill_u32<<<ceil_div(ndst, 256), 256, 0, s>>>(W.mcode, 0u, ndst);
  k_fill_u32<<<ceil_div(ndst, 256), 256, 0, s>>>((unsigned*)W.ssum, 0u, ndst);
  k_fill_u32<<<ceil_div((long)ndst * HDIM, 256), 256, 0, s>>>((unsigned*)W.agg, 0u,
                                                              (long)ndst * HDIM);
  k_fill_u32<<<1, 64, 0, s>>>((unsigned*)W.scal, 0u, 16);
  if (self_loops) k_sum_attr<<<256, 256, 0, s>>>(eattr, E, W.scal);
  k_layer_scalars<<<1, 128, 0, s>>>(lin_e, att_e, W.scal, self_loops ? (1.0f / E) : 0.0f);
  int nloops = self_loops ? ndst : 0;
  int total = E + nloops;
  k_edge_alpha_max<<<ceil_div(total, 256), 256, 0, s>>>(esrc, edst, eattr, E, nloops, W.asrc,
                                                        att_dst ? W.adst : nullptr, W.scal,
                                                        W.alpha, W.mcode);
  k_node_decode<<<ceil_div(ndst, 256), 256, 0, s>>>(W.mcode, W.mval, ndst);
  k_edge_exp_sum<<<ceil_div(total, 256), 256, 0, s>>>(edst, E, nloops, W.mval, W.alpha, W.ssum);
  k_edge_agg<<<ceil_div(total, 8), 256, 0, s>>>(esrc, edst, E, nloops, W.alpha, W.ssum,
                                                W.h_lin, W.agg);
  k_finalize<<<ceil_div((long)ndst * HDIM, 256), 256, 0, s>>>(W.agg, bias, out,
                                                              (long)ndst * HDIM, HDIM,
                                                              relu_out ? 1 : 0);
}

extern "C" void kernel_launch(void* const* d_in, const int* in_sizes, int n_in,
                              void* d_out, int out_size, void* d_ws, size_t ws_size,
                              hipStream_t stream) {
  (void)in_sizes; (void)n_in; (void)out_size; (void)ws_size;
  // inputs in setup_inputs() order
  const float* spatial_x      = (const float*)d_in[0];
  const float* spatial_global = (const float*)d_in[1];
  const float* species_x      = (const float*)d_in[2];
  const float* species_phylo  = (const float*)d_in[3];
  const int*   ss_src = (const int*)d_in[4];
  const int*   ss_dst = (const int*)d_in[5];
  const float* ss_attr = (const float*)d_in[6];
  const int*   bp_src = (const int*)d_in[7];
  const int*   bp_dst = (const int*)d_in[8];
  const float* bp_attr = (const float*)d_in[9];
  const int*   pp_src = (const int*)d_in[10];
  const int*   pp_dst = (const int*)d_in[11];
  const float* pp_attr = (const float*)d_in[12];
  const float* sg0_lin = (const float*)d_in[13];
  const float* sg0_att_src = (const float*)d_in[14];
  const float* sg0_att_dst = (const float*)d_in[15];
  const float* sg0_lin_e = (const float*)d_in[16];
  const float* sg0_att_e = (const float*)d_in[17];
  const float* sg0_bias = (const float*)d_in[18];
  const float* sg1_lin = (const float*)d_in[19];
  const float* sg1_att_src = (const float*)d_in[20];
  const float* sg1_att_dst = (const float*)d_in[21];
  const float* sg1_lin_e = (const float*)d_in[22];
  const float* sg1_att_e = (const float*)d_in[23];
  const float* sg1_bias = (const float*)d_in[24];
  const float* bpc_lin = (const float*)d_in[25];
  const float* bpc_att_src = (const float*)d_in[26];
  const float* bpc_lin_e = (const float*)d_in[27];
  const float* bpc_att_e = (const float*)d_in[28];
  const float* bpc_bias = (const float*)d_in[29];
  const float* pg0_lin = (const float*)d_in[30];
  const float* pg0_att_src = (const float*)d_in[31];
  const float* pg0_att_dst = (const float*)d_in[32];
  const float* pg0_lin_e = (const float*)d_in[33];
  const float* pg0_att_e = (const float*)d_in[34];
  const float* pg0_bias = (const float*)d_in[35];
  const float* pg1_lin = (const float*)d_in[36];
  const float* pg1_att_src = (const float*)d_in[37];
  const float* pg1_att_dst = (const float*)d_in[38];
  const float* pg1_lin_e = (const float*)d_in[39];
  const float* pg1_att_e = (const float*)d_in[40];
  const float* pg1_bias = (const float*)d_in[41];
  const float* fc_w = (const float*)d_in[42];
  const float* fc_b = (const float*)d_in[43];

  // workspace bump allocator
  char* base = (char*)d_ws;
  size_t off = 0;
  auto take = [&](size_t bytes) -> void* {
    void* p = base + off;
    off = (off + bytes + 255) & ~(size_t)255;
    return p;
  };
  WS W;
  W.h_cur = (float*)take((size_t)NSP * HDIM * 4);
  W.h_lin = (float*)take((size_t)NSP * HDIM * 4);
  W.agg   = (float*)take((size_t)NSP * HDIM * 4);
  W.xf16  = (_Float16*)take((size_t)NSP * 224 * 2);
  W.wf16  = (_Float16*)take((size_t)HDIM * 224 * 2);
  W.asrc  = (float*)take((size_t)NSP * 4);
  W.adst  = (float*)take((size_t)NSP * 4);
  W.alpha = (float*)take((size_t)(ESS + NSP) * 4);
  W.mcode = (unsigned*)take((size_t)NSP * 4);
  W.mval  = (float*)take((size_t)NSP * 4);
  W.ssum  = (float*)take((size_t)NSP * 4);
  W.scal  = (float*)take(64);

  hipStream_t s = stream;

  // space GNN layer 0: x = concat(spatial_x[16], spatial_global[4]) -> K padded to 32
  gat_layer(s, W, spatial_x, 16, spatial_global, 4, nullptr, 0, NSP, 32,
            sg0_lin, sg0_att_src, sg0_att_dst, sg0_lin_e, sg0_att_e, sg0_bias,
            ss_src, ss_dst, ss_attr, ESS, NSP, true, true, W.h_cur);
  // space GNN layer 1
  gat_layer(s, W, W.h_cur, HDIM, nullptr, 0, nullptr, 0, NSP, HDIM,
            sg1_lin, sg1_att_src, sg1_att_dst, sg1_lin_e, sg1_att_e, sg1_bias,
            ss_src, ss_dst, ss_attr, ESS, NSP, true, true, W.h_cur);
  // bipartite spatial->species (no dst attention, no self loops); result in h_cur[0..NSPEC)
  gat_layer(s, W, W.h_cur, HDIM, nullptr, 0, nullptr, 0, NSP, HDIM,
            bpc_lin, bpc_att_src, nullptr, bpc_lin_e, bpc_att_e, bpc_bias,
            bp_src, bp_dst, bp_attr, EBP, NSPEC, false, true, W.h_cur);
  // species GNN layer 0: x = concat(s2s[128], species_x[32], species_phylo[64]) -> K=224
  gat_layer(s, W, W.h_cur, HDIM, species_x, 32, species_phylo, 64, NSPEC, 224,
            pg0_lin, pg0_att_src, pg0_att_dst, pg0_lin_e, pg0_att_e, pg0_bias,
            pp_src, pp_dst, pp_attr, EPP, NSPEC, true, true, W.h_cur);
  // species GNN layer 1
  gat_layer(s, W, W.h_cur, HDIM, nullptr, 0, nullptr, 0, NSPEC, HDIM,
            pg1_lin, pg1_att_src, pg1_att_dst, pg1_lin_e, pg1_att_e, pg1_bias,
            pp_src, pp_dst, pp_attr, EPP, NSPEC, true, true, W.h_cur);

  // final FC: out = g @ fc_w.T + fc_b  (WMMA, single 16-wide N tile)
  long gx = (long)NSPEC * HDIM;
  k_concat_f16<<<ceil_div(gx, 256), 256, 0, s>>>(W.h_cur, HDIM, nullptr, 0, nullptr, 0,
                                                 W.xf16, HDIM, gx);
  k_concat_f16<<<ceil_div((long)16 * HDIM, 256), 256, 0, s>>>(fc_w, HDIM, nullptr, 0, nullptr, 0,
                                                              W.wf16, HDIM, (long)16 * HDIM);
  dim3 gf(NSPEC / 16, 1);
  k_gemm_wmma<1><<<gf, 32, 0, s>>>(W.xf16, W.wf16, W.h_lin, NSPEC, 16, HDIM);
  k_finalize<<<ceil_div((long)NSPEC * 16, 256), 256, 0, s>>>(W.h_lin, fc_b, (float*)d_out,
                                                             (long)NSPEC * 16, 16, 0);
}
